// registration_net_27187142983928
// MI455X (gfx1250) — matrix operand order repaired and checked by reading, not compiled, not run on hardware
//
#include <hip/hip_runtime.h>
#include <math.h>

typedef __bf16 bf16;
typedef __bf16 v16bf __attribute__((ext_vector_type(16)));
typedef float  v8f   __attribute__((ext_vector_type(8)));
typedef int    v4i   __attribute__((ext_vector_type(4)));
typedef __attribute__((address_space(1))) v4i g1v4i;   // global 16B vector
typedef __attribute__((address_space(3))) v4i l3v4i;   // LDS 16B vector

#if defined(__gfx1250__) && __has_builtin(__builtin_amdgcn_global_load_async_to_lds_b128) && \
    __has_builtin(__builtin_amdgcn_s_wait_asynccnt)
#define RN_HAS_ASYNC 1
#else
#define RN_HAS_ASYNC 0
#endif

#define LRELU 0.1f
#define GSLOOP(i, total) for (long long i = (long long)blockIdx.x * blockDim.x + threadIdx.x; \
                              i < (total); i += (long long)gridDim.x * blockDim.x)

static __device__ __forceinline__ float b2f(bf16 v) { return (float)v; }
static __device__ __forceinline__ bf16  f2b(float v) { return (bf16)v; }

// ---------------------------------------------------------------------------
// Weight packing: OIHW fp32 -> bf16 fragments pre-swizzled to the CDNA5 WMMA
// A-matrix 16x32 operand, in this kernel's contiguous k order (not the ISA table order).  Block layout:
//   wpk[((coTile*9 + kp)*nCC + cc)*512 + lane*16 + e]
// lane<16 : co=tile*16+lane,    Klocal = e
// lane>=16: co=tile*16+lane-16, Klocal = e + 16
// ---------------------------------------------------------------------------
__global__ void rn_pack_w(const float* __restrict__ w, const float* __restrict__ bsrc,
                          bf16* __restrict__ wpk, float* __restrict__ bdst,
                          int Cin, int Cout, int CinP, int CoutP) {
  const int nCC = CinP >> 5;
  const long long total = (long long)(CoutP >> 4) * 9 * nCC * 512;
  GSLOOP(i, total) {
    int e    = (int)(i & 15);
    int lane = (int)((i >> 4) & 31);
    long long blk = i >> 9;
    int cc   = (int)(blk % nCC);
    long long t2 = blk / nCC;
    int kp   = (int)(t2 % 9);
    int tile = (int)(t2 / 9);
    int co = tile * 16 + (lane & 15);
    int Kl = e + ((lane >> 4) << 4);
    int ci = cc * 32 + Kl;
    float val = 0.f;
    if (co < Cout && ci < Cin) {
      int ky = kp / 3, kx = kp - ky * 3;
      val = w[(((long long)co * Cin + ci) * 3 + ky) * 3 + kx];
    }
    wpk[i] = f2b(val);
  }
  GSLOOP(c, (long long)CoutP) bdst[c] = (c < Cout) ? bsrc[c] : 0.f;
}

// NCHW fp32 -> NHWC bf16 (channel-padded)
__global__ void rn_cast(const float* __restrict__ in, bf16* __restrict__ out,
                        int B, int C, int H, int W, int Cp) {
  GSLOOP(i, (long long)B * H * W * Cp) {
    int c = (int)(i % Cp); long long p = i / Cp;
    int x = (int)(p % W); p /= W;
    int y = (int)(p % H); int b = (int)(p / H);
    float v = (c < C) ? in[(((long long)b * C + c) * H + y) * W + x] : 0.f;
    out[i] = f2b(v);
  }
}

// ---------------------------------------------------------------------------
// Implicit-GEMM 3x3 dilated conv, bf16 WMMA, f32 accum, fused bias+leakyReLU.
// Block = 8 waves: 4 co-tiles x 2 pixel-halves. 32 output pixels per block.
// Input slab is staged into LDS via GLOBAL_LOAD_ASYNC_TO_LDS (ASYNCcnt).
// ---------------------------------------------------------------------------
#define TW 32
#define LDS_ELEMS 15360  // >= 3 * max(LW * CinP)

__global__ __launch_bounds__(256) void rn_conv3x3_wmma(
    const bf16* __restrict__ in, const bf16* __restrict__ wpk,
    const float* __restrict__ bias, bf16* __restrict__ out,
    int Hin, int Win, int CinP, int CoutP, int stride, int dil, int relu,
    int coGroups) {
  __shared__ bf16 lds[LDS_ELEMS];
  const int Wout = Win / stride, Hout = Hin / stride;
  const int LW = TW * stride + 2 * dil;
  const int tid = threadIdx.x, lane = tid & 31, wid = tid >> 5;
  const int x0 = blockIdx.x * TW;
  const int y  = blockIdx.y;
  const int b  = blockIdx.z / coGroups;
  const int cg = blockIdx.z % coGroups;

  // ---- stage 3 input rows (y*stride + (r-1)*dil) into LDS, zero-padded ----
  const int gx0 = x0 * stride - dil;
  const long long inBase = (long long)b * Hin * Win * CinP;
  const int c8PerRow = CinP >> 3;
  const int vecPerRow = LW * c8PerRow;
  for (int idx = tid; idx < 3 * vecPerRow; idx += 256) {
    int r = idx / vecPerRow;
    int rem = idx - r * vecPerRow;
    int col = rem / c8PerRow;
    int c8 = rem - col * c8PerRow;
    int gy = y * stride + (r - 1) * dil;
    int gx = gx0 + col;
    bf16* lp = lds + ((long long)r * LW + col) * CinP + c8 * 8;
    if (gy >= 0 && gy < Hin && gx >= 0 && gx < Win) {
      const bf16* gp = in + inBase + ((long long)gy * Win + gx) * CinP + c8 * 8;
#if RN_HAS_ASYNC
      // VGPR-free 16B/lane copy, tracked by ASYNCcnt
      __builtin_amdgcn_global_load_async_to_lds_b128(
          (g1v4i*)(unsigned long long)(const void*)gp,
          (l3v4i*)(unsigned)(unsigned long long)(void*)lp,
          0, 0);
#else
      *(uint4*)lp = *(const uint4*)gp;
#endif
    } else {
      *(uint4*)lp = make_uint4(0u, 0u, 0u, 0u);
    }
  }
  // warm L2 for the next output row's fresh bottom input row
  {
    int gy = (y + 1) * stride + dil;
    if (gy < Hin)
      for (int col = tid; col < LW; col += 256) {
        int gx = gx0 + col;
        if (gx >= 0 && gx < Win)
          __builtin_prefetch(in + inBase + ((long long)gy * Win + gx) * CinP, 0, 1);
      }
  }
#if RN_HAS_ASYNC
  __builtin_amdgcn_s_wait_asynccnt(0);
#endif
  __syncthreads();

  const int coTile = cg * 4 + (wid & 3);
  const int px0 = (wid >> 2) * 16;
  if (coTile * 16 < CoutP) {            // wave-uniform guard: EXEC stays all-1
    v8f acc = {};
    const int nCC = CinP >> 5;
    const bf16* wbase = wpk + (long long)coTile * 9 * nCC * 512 + lane * 16;
    const int bcol = (px0 + (lane & 15)) * stride;
    const int ksel = (lane >> 4) * 16;
    for (int kp = 0; kp < 9; ++kp) {
      const int ky = kp / 3, kx = kp - ky * 3;
      const bf16* lb = lds + ((long long)ky * LW + bcol + kx * dil) * CinP + ksel;
      const bf16* ab = wbase + (long long)kp * nCC * 512;
      for (int cc = 0; cc < nCC; ++cc) {
        v16bf a  = *(const v16bf*)(ab + (long long)cc * 512);
        v16bf bm = *(const v16bf*)(lb + cc * 32);
        acc = __builtin_amdgcn_wmma_f32_16x16x32_bf16(false, a, false, bm,
                                                      (short)0, acc, false, false);
      }
    }
    // D layout: vgpr v, lane l -> co = coTile*16 + v + 8*(l>=16), px = l&15
    const int px = px0 + (lane & 15);
    const int coB = coTile * 16 + (lane >> 4) * 8;
    bf16 o[8];
#pragma unroll
    for (int v = 0; v < 8; ++v) {
      float r = acc[v] + bias[coB + v];
      if (relu) r = r > 0.f ? r : r * LRELU;
      o[v] = f2b(r);
    }
    *(uint4*)(out + (((long long)b * Hout + y) * Wout + x0 + px) * CoutP + coB) =
        *(const uint4*)o;
  }
}

// ---- per-channel mean/var (instance: per (b,c); batch: per c) -------------
__global__ void rn_stats(const bf16* __restrict__ x, float* __restrict__ mean,
                         float* __restrict__ var, int B, int H, int W, int Cp,
                         int C, int instanceMode) {
  __shared__ float ssum[256], ssq[256];
  const int c = threadIdx.x % C, s = threadIdx.x / C, S = 256 / C;
  const long long npix = (long long)H * W;
  float sum = 0.f, sq = 0.f;
  const int b0 = instanceMode ? (int)blockIdx.x : 0;
  const int b1 = instanceMode ? (int)blockIdx.x + 1 : B;
  for (int b = b0; b < b1; ++b)
    for (long long p = s; p < npix; p += S) {
      float v = b2f(x[((long long)b * npix + p) * Cp + c]);
      sum += v; sq += v * v;
    }
  ssum[threadIdx.x] = sum; ssq[threadIdx.x] = sq;
  __syncthreads();
  if (s == 0) {
    for (int k = 1; k < S; ++k) { sum += ssum[k * C + c]; sq += ssq[k * C + c]; }
    float n = (float)((long long)(b1 - b0) * npix);
    float m = sum / n, v = sq / n - m * m;
    int oi = instanceMode ? (int)blockIdx.x * C + c : c;
    mean[oi] = m; var[oi] = v < 0.f ? 0.f : v;
  }
}

__global__ void rn_norm_act(const bf16* __restrict__ x, const float* __restrict__ mean,
                            const float* __restrict__ var, const bf16* __restrict__ resid,
                            bf16* __restrict__ out, int B, long long HW, int Cp, int C,
                            int instanceMode, int relu) {
  GSLOOP(i, (long long)B * HW * Cp) {
    int c = (int)(i % Cp);
    int b = (int)((i / Cp) / HW);
    float v = b2f(x[i]);
    if (c < C) {
      int oi = instanceMode ? b * C + c : c;
      v = (v - mean[oi]) * rsqrtf(var[oi] + 1e-5f);
    } else v = 0.f;
    if (resid) v += b2f(resid[i]);
    if (relu) v = v > 0.f ? v : v * LRELU;
    out[i] = f2b(v);
  }
}

// ---- separable Gaussian blur, weights generated in-register ---------------
template <typename T>
__global__ void rn_blur_h(const T* __restrict__ in, T* __restrict__ out,
                          int B, int H, int W, int Cp, int ks, float sigma, int edge) {
  const float r = (ks - 1) * 0.5f, inv2s2 = 1.f / (2.f * sigma * sigma);
  GSLOOP(i, (long long)B * H * W * Cp) {
    int c = (int)(i % Cp); long long p = i / Cp;
    int x = (int)(p % W); long long q = p / W;
    float acc = 0.f, wsum = 0.f;
    for (int t = 0; t < ks; ++t) {
      float wt = __expf(-(t - r) * (t - r) * inv2s2); wsum += wt;
      int xx = x + t - ks / 2;
      bool valid = edge || (xx >= 0 && xx < W);
      xx = xx < 0 ? 0 : (xx > W - 1 ? W - 1 : xx);
      if (valid) acc += wt * (float)in[(q * W + xx) * Cp + c];
    }
    out[i] = (T)(acc / wsum);
  }
}

template <typename T>
__global__ void rn_blur_v(const T* __restrict__ in, T* __restrict__ out,
                          int B, int H, int W, int Cp, int ks, float sigma, int edge) {
  const float r = (ks - 1) * 0.5f, inv2s2 = 1.f / (2.f * sigma * sigma);
  GSLOOP(i, (long long)B * H * W * Cp) {
    int c = (int)(i % Cp); long long p = i / Cp;
    int x = (int)(p % W); long long q = p / W;
    int y = (int)(q % H); int b = (int)(q / H);
    float acc = 0.f, wsum = 0.f;
    for (int t = 0; t < ks; ++t) {
      float wt = __expf(-(t - r) * (t - r) * inv2s2); wsum += wt;
      int yy = y + t - ks / 2;
      bool valid = edge || (yy >= 0 && yy < H);
      yy = yy < 0 ? 0 : (yy > H - 1 ? H - 1 : yy);
      if (valid) acc += wt * (float)in[(((long long)b * H + yy) * W + x) * Cp + c];
    }
    out[i] = (T)(acc / wsum);
  }
}

// ---- 7x7 dilated local cost volume: mean_c (f2 - shift(f1s))^2 ------------
__global__ void rn_corr(const bf16* __restrict__ f1s, const bf16* __restrict__ f2,
                        bf16* __restrict__ corr, int B, int H, int W, int C,
                        int chOff, int CpOut) {
  GSLOOP(i, (long long)B * H * W) {
    int x = (int)(i % W); long long t = i / W;
    int y = (int)(t % H); int b = (int)(t / H);
    const bf16* f2p = f2 + i * C;           // f2 tensors are Cp==C==64
    const bf16* f1b = f1s + (long long)b * H * W * C;
    bf16* op = corr + i * CpOut + chOff;
    for (int s = 0; s < 49; ++s) {
      int dy = (s / 7) * 4 - 12, dx = (s % 7) * 4 - 12;
      int yy = y + dy, xx = x + dx;
      bool valid = (yy >= 0 && yy < H && xx >= 0 && xx < W);
      const bf16* f1p = valid ? f1b + ((long long)yy * W + xx) * C : nullptr;
      float acc = 0.f;
      for (int c = 0; c < 64; ++c) {
        float d = b2f(f2p[c]) - (valid ? b2f(f1p[c]) : 0.f);
        acc += d * d;
      }
      op[s] = f2b(acc * (1.f / 64.f));
    }
  }
}

// ---- bilinear grid-sample warp (disp in normalized units, zero outside) ---
__global__ void rn_warp(const bf16* __restrict__ f, const float* __restrict__ disp,
                        bf16* __restrict__ out, int B, int H, int W, int Cp) {
  GSLOOP(i, (long long)B * H * W) {
    int x = (int)(i % W); long long t = i / W;
    int y = (int)(t % H); int b = (int)(t / H);
    float gx = (W > 1 ? -1.f + 2.f * x / (W - 1) : 0.f) + disp[i * 2 + 0];
    float gy = (H > 1 ? -1.f + 2.f * y / (H - 1) : 0.f) + disp[i * 2 + 1];
    float px = ((gx + 1.f) * W - 1.f) * 0.5f;
    float py = ((gy + 1.f) * H - 1.f) * 0.5f;
    float x0f = floorf(px), y0f = floorf(py);
    int xi[2] = {(int)x0f, (int)x0f + 1}, yi[2] = {(int)y0f, (int)y0f + 1};
    float wx[2] = {1.f - (px - x0f), px - x0f};
    float wy[2] = {1.f - (py - y0f), py - y0f};
    const bf16* base = f + (long long)b * H * W * Cp;
    const bf16* sp[4]; float sw[4];
    for (int k = 0; k < 4; ++k) {
      int xx = xi[k & 1], yy = yi[k >> 1];
      bool valid = (xx >= 0 && xx <= W - 1 && yy >= 0 && yy <= H - 1);
      int xc = xx < 0 ? 0 : (xx > W - 1 ? W - 1 : xx);
      int yc = yy < 0 ? 0 : (yy > H - 1 ? H - 1 : yy);
      sp[k] = base + ((long long)yc * W + xc) * Cp;
      sw[k] = valid ? wx[k & 1] * wy[k >> 1] : 0.f;
    }
    bf16* op = out + i * Cp;
    for (int c = 0; c < Cp; ++c)
      op[c] = f2b(sw[0] * b2f(sp[0][c]) + sw[1] * b2f(sp[1][c]) +
                  sw[2] * b2f(sp[2][c]) + sw[3] * b2f(sp[3][c]));
  }
}

// ---- half-pixel bilinear resize of fp32 2-channel NHWC --------------------
__global__ void rn_resize2(const float* __restrict__ in, float* __restrict__ out,
                           int B, int Hi, int Wi, int Ho, int Wo) {
  GSLOOP(i, (long long)B * Ho * Wo * 2) {
    int c = (int)(i & 1); long long t = i >> 1;
    int x = (int)(t % Wo); t /= Wo;
    int y = (int)(t % Ho); int b = (int)(t / Ho);
    float sx = ((x + 0.5f) * Wi) / Wo - 0.5f;
    float sy = ((y + 0.5f) * Hi) / Ho - 0.5f;
    sx = fminf(fmaxf(sx, 0.f), (float)(Wi - 1));
    sy = fminf(fmaxf(sy, 0.f), (float)(Hi - 1));
    int x0 = (int)sx, y0 = (int)sy;
    int x1 = x0 + 1 < Wi ? x0 + 1 : Wi - 1;
    int y1 = y0 + 1 < Hi ? y0 + 1 : Hi - 1;
    float fx = sx - x0, fy = sy - y0;
    const float* bp = in + (long long)b * Hi * Wi * 2;
    float v00 = bp[((long long)y0 * Wi + x0) * 2 + c], v01 = bp[((long long)y0 * Wi + x1) * 2 + c];
    float v10 = bp[((long long)y1 * Wi + x0) * 2 + c], v11 = bp[((long long)y1 * Wi + x1) * 2 + c];
    out[i] = (v00 * (1 - fx) + v01 * fx) * (1 - fy) + (v10 * (1 - fx) + v11 * fx) * fy;
  }
}

__global__ void rn_copy_ch(const bf16* __restrict__ in, bf16* __restrict__ out,
                           long long npix, int CpIn, int CpOut, int nCh,
                           int srcOff, int dstOff) {
  GSLOOP(i, npix * nCh) {
    int c = (int)(i % nCh); long long p = i / nCh;
    out[p * CpOut + dstOff + c] = in[p * CpIn + srcOff + c];
  }
}

__global__ void rn_fill0(bf16* __restrict__ p, long long n) { GSLOOP(i, n) p[i] = f2b(0.f); }

__global__ void rn_b2f2(const bf16* __restrict__ in, float* __restrict__ out,
                        long long npix, int CpIn) {
  GSLOOP(i, npix * 2) out[i] = b2f(in[(i >> 1) * CpIn + (i & 1)]);
}

__global__ void rn_f2b2(const float* __restrict__ in, bf16* __restrict__ out,
                        long long npix, int CpOut, int dstOff) {
  GSLOOP(i, npix * 2) out[(i >> 1) * CpOut + dstOff + (i & 1)] = f2b(in[i]);
}

// out = clamp(a*sa + b*sb, lo, hi), per-channel scales, null ptr -> 0
__global__ void rn_comb(const float* __restrict__ a, const float* __restrict__ b,
                        float* __restrict__ out, long long npix,
                        float sax, float say, float sbx, float sby,
                        float lo, float hi) {
  GSLOOP(i, npix * 2) {
    int c = (int)(i & 1);
    float v = (a ? a[i] : 0.f) * (c ? say : sax) + (b ? b[i] : 0.f) * (c ? sby : sbx);
    out[i] = fminf(fmaxf(v, lo), hi);
  }
}

__global__ void rn_out_nchw(const float* __restrict__ in, float* __restrict__ out,
                            int B, int H, int W) {
  GSLOOP(i, (long long)B * 2 * H * W) {
    int x = (int)(i % W); long long t = i / W;
    int y = (int)(t % H); t /= H;
    int c = (int)(t % 2); int b = (int)(t / 2);
    out[i] = in[(((long long)b * H + y) * W + x) * 2 + c];
  }
}

// ===========================================================================
// Host orchestration
// ===========================================================================
struct ConvW { bf16* wpk; float* bias; int Cin, Cout, CinP, CoutP; };

static inline unsigned nblk(long long n) {
  long long b = (n + 255) >> 8; if (b < 1) b = 1; if (b > 32768) b = 32768;
  return (unsigned)b;
}
static inline void* aalloc(char*& p, size_t bytes) {
  void* r = p; p += (bytes + 255) & ~(size_t)255; return r;
}
static inline int rup32(int c) { return (c + 31) & ~31; }

static void run_conv(hipStream_t st, const bf16* in, int B, int Hin, int Win,
                     const ConvW& cw, bf16* out, int stride, int dil, int relu) {
  int groups = (cw.CoutP / 16 + 3) / 4;
  dim3 g((Win / stride) / TW, Hin / stride, B * groups);
  rn_conv3x3_wmma<<<g, 256, 0, st>>>(in, cw.wpk, cw.bias, out, Hin, Win,
                                     cw.CinP, cw.CoutP, stride, dil, relu, groups);
}

static void run_norm(hipStream_t st, char*& ar, bf16* x, const bf16* resid,
                     int B, int H, int W, int Cp, int C, int inst, int relu) {
  int nstat = inst ? B * C : C;
  float* mean = (float*)aalloc(ar, nstat * 4);
  float* var  = (float*)aalloc(ar, nstat * 4);
  rn_stats<<<inst ? B : 1, 256, 0, st>>>(x, mean, var, B, H, W, Cp, C, inst);
  long long n = (long long)B * H * W * Cp;
  rn_norm_act<<<nblk(n), 256, 0, st>>>(x, mean, var, resid, x, B,
                                       (long long)H * W, Cp, C, inst, relu);
}

template <typename T>
static void run_blur(hipStream_t st, const T* in, T* tmp, T* out, int B, int H,
                     int W, int Cp, int ks, float sig, int edge) {
  long long n = (long long)B * H * W * Cp;
  rn_blur_h<T><<<nblk(n), 256, 0, st>>>(in, tmp, B, H, W, Cp, ks, sig, edge);
  rn_blur_v<T><<<nblk(n), 256, 0, st>>>(tmp, out, B, H, W, Cp, ks, sig, edge);
}

// est-head: 3x (conv + BN + lrelu) with dil 1,2,4, then conv -> 2ch (Cp=32)
static bf16* run_est(hipStream_t st, char*& ar, const ConvW* CW, int kb,
                     const bf16* x, int h, int w) {
  size_t hw = (size_t)h * w;
  const int dils[3] = {1, 2, 4};
  const bf16* cur = x;
  for (int l = 0; l < 3; ++l) {
    const ConvW& c = CW[kb + l];
    bf16* o = (bf16*)aalloc(ar, 2 * hw * c.CoutP * 2);
    run_conv(st, cur, 2, h, w, c, o, 1, dils[l], 0);
    run_norm(st, ar, o, nullptr, 2, h, w, c.CoutP, c.Cout, /*inst=*/0, /*relu=*/1);
    cur = o;
  }
  const ConvW& c4 = CW[kb + 3];
  bf16* e4 = (bf16*)aalloc(ar, 2 * hw * c4.CoutP * 2);
  run_conv(st, cur, 2, h, w, c4, e4, 1, 1, 0);
  return e4;
}

// disp_estimator: returns normalized 2ch fp32 disparity (2,h,w,2)
static void run_matcher(hipStream_t st, char*& ar, const ConvW* CW, int kb,
                        const bf16* cat4, int h, int w, int preDil, float* dispOut) {
  size_t hw = (size_t)h * w;
  bf16* feat = (bf16*)aalloc(ar, 4 * hw * 64 * 2);
  run_conv(st, cat4, 4, h, w, CW[kb], feat, 1, preDil, 0);

  bf16* cat2 = (bf16*)aalloc(ar, 2 * hw * 128 * 2);
  rn_copy_ch<<<nblk(2 * hw * 64), 256, 0, st>>>(feat, cat2, 2 * hw, 64, 128, 64, 0, 0);
  rn_copy_ch<<<nblk(2 * hw * 64), 256, 0, st>>>(feat + 2 * hw * 64, cat2, 2 * hw, 64, 128, 64, 0, 64);
  bf16* fcb = (bf16*)aalloc(ar, 2 * hw * 128 * 2);
  run_conv(st, cat2, 2, h, w, CW[kb + 1], fcb, 1, 1, 1);
  bf16* fco = (bf16*)aalloc(ar, 2 * hw * 64 * 2);
  run_conv(st, fcb, 2, h, w, CW[kb + 2], fco, 1, 1, 0);

  bf16* btmp = (bf16*)aalloc(ar, 2 * hw * 64 * 2);
  bf16* f1s  = (bf16*)aalloc(ar, 2 * hw * 64 * 2);
  run_blur<bf16>(st, feat, btmp, f1s, 2, h, w, 64, 13, 3.0f, /*edge=*/0);

  bf16* corrb = (bf16*)aalloc(ar, 2 * hw * 128 * 2);
  rn_fill0<<<nblk(2 * hw * 128), 256, 0, st>>>(corrb, (long long)2 * hw * 128);
  rn_copy_ch<<<nblk(2 * hw * 64), 256, 0, st>>>(fco, corrb, 2 * hw, 64, 128, 64, 0, 0);
  rn_corr<<<nblk(2 * hw), 256, 0, st>>>(f1s, feat + 2 * hw * 64, corrb, 2, h, w, 64, 64, 128);

  bf16* e4 = run_est(st, ar, CW, kb + 3, corrb, h, w);

  float* draw  = (float*)aalloc(ar, 2 * hw * 2 * 4);
  float* dtmp  = (float*)aalloc(ar, 2 * hw * 2 * 4);
  float* dblur = (float*)aalloc(ar, 2 * hw * 2 * 4);
  rn_b2f2<<<nblk(2 * hw * 2), 256, 0, st>>>(e4, draw, 2 * hw, 32);
  run_blur<float>(st, draw, dtmp, dblur, 2, h, w, 2, 13, 3.0f, /*edge=*/1);
  rn_comb<<<nblk(2 * hw * 2), 256, 0, st>>>(dblur, nullptr, dtmp, 2 * hw,
                                            1.f, 1.f, 0.f, 0.f, -300.f, 300.f);
  rn_comb<<<nblk(2 * hw * 2), 256, 0, st>>>(dtmp, nullptr, dispOut, 2 * hw,
                                            1.f / (w - 1), 1.f / (h - 1), 0.f, 0.f,
                                            -1e30f, 1e30f);
}

// disp_refiner: dispOut = scaleup + delta  (both (2,h,w,2) fp32, pixel units)
static void run_refiner(hipStream_t st, char*& ar, const ConvW* CW,
                        const bf16* cat4, int h, int w, const float* scaleup,
                        float* dispOut) {
  size_t hw = (size_t)h * w;
  bf16* feat = (bf16*)aalloc(ar, 4 * hw * 64 * 2);
  run_conv(st, cat4, 4, h, w, CW[28], feat, 1, 1, 0);
  bf16* cat2 = (bf16*)aalloc(ar, 2 * hw * 128 * 2);
  rn_copy_ch<<<nblk(2 * hw * 64), 256, 0, st>>>(feat, cat2, 2 * hw, 64, 128, 64, 0, 0);
  rn_copy_ch<<<nblk(2 * hw * 64), 256, 0, st>>>(feat + 2 * hw * 64, cat2, 2 * hw, 64, 128, 64, 0, 64);
  bf16* fcb = (bf16*)aalloc(ar, 2 * hw * 128 * 2);
  run_conv(st, cat2, 2, h, w, CW[29], fcb, 1, 1, 1);
  bf16* fco = (bf16*)aalloc(ar, 2 * hw * 64 * 2);
  run_conv(st, fcb, 2, h, w, CW[30], fco, 1, 1, 0);

  bf16* corrb = (bf16*)aalloc(ar, 2 * hw * 96 * 2);  // 64 + 2 -> Cp 96
  rn_fill0<<<nblk(2 * hw * 96), 256, 0, st>>>(corrb, (long long)2 * hw * 96);
  rn_copy_ch<<<nblk(2 * hw * 64), 256, 0, st>>>(fco, corrb, 2 * hw, 64, 96, 64, 0, 0);
  rn_f2b2<<<nblk(2 * hw * 2), 256, 0, st>>>(scaleup, corrb, 2 * hw, 96, 64);

  bf16* e4 = run_est(st, ar, CW, 31, corrb, h, w);
  float* delta = (float*)aalloc(ar, 2 * hw * 2 * 4);
  rn_b2f2<<<nblk(2 * hw * 2), 256, 0, st>>>(e4, delta, 2 * hw, 32);
  rn_comb<<<nblk(2 * hw * 2), 256, 0, st>>>(scaleup, delta, dispOut, 2 * hw,
                                            1.f, 1.f, 1.f, 1.f, -1e30f, 1e30f);
}

// conv index table (flatten order of params dict)
static const int kCin[35] = {64,64,64,64, 64,64,64,64, 64,64, 64,64, 64,64,
                             64, 128,128, 113,32,16,8,
                             64, 128,128, 113,32,16,8,
                             64, 128,128, 66,32,16,8};
static const int kCout[35] = {64,64,64,64, 64,64,64,64, 64,64, 64,64, 64,64,
                              64, 128,64, 32,16,8,2,
                              64, 128,64, 32,16,8,2,
                              64, 128,64, 32,16,8,2};

extern "C" void kernel_launch(void* const* d_in, const int* in_sizes, int n_in,
                              void* d_out, int out_size, void* d_ws, size_t ws_size,
                              hipStream_t stream) {
  (void)in_sizes; (void)n_in; (void)out_size; (void)ws_size;
  hipStream_t st = stream;
  char* ar = (char*)d_ws;

  // ---- pack all conv weights to WMMA fragment layout ----
  ConvW CW[35];
  for (int k = 0; k < 35; ++k) {
    ConvW& c = CW[k];
    c.Cin = kCin[k]; c.Cout = kCout[k];
    c.CinP = rup32(c.Cin); c.CoutP = rup32(c.Cout);
    size_t we = (size_t)c.CoutP * 9 * c.CinP;
    c.wpk  = (bf16*)aalloc(ar, we * 2);
    c.bias = (float*)aalloc(ar, c.CoutP * 4);
    rn_pack_w<<<nblk((long long)we), 256, 0, st>>>(
        (const float*)d_in[2 + 2 * k], (const float*)d_in[3 + 2 * k],
        c.wpk, c.bias, c.Cin, c.Cout, c.CinP, c.CoutP);
  }

  const int H = 256, W = 256;
  const size_t hwF = (size_t)H * W;          // 256^2
  // ---- cast inputs to NHWC bf16, batch-concat [src|tgt] -> (4,256,256,64) --
  bf16* A = (bf16*)aalloc(ar, 4 * hwF * 64 * 2);
  bf16* Bb = (bf16*)aalloc(ar, 4 * hwF * 64 * 2);
  rn_cast<<<nblk(2 * hwF * 64), 256, 0, st>>>((const float*)d_in[0], A, 2, 64, H, W, 64);
  rn_cast<<<nblk(2 * hwF * 64), 256, 0, st>>>((const float*)d_in[1], A + 2 * hwF * 64, 2, 64, H, W, 64);

  // ---- unshare: 4 res-blocks, dil {1,2,2,4}, un1 on batch 0-1, un2 on 2-3 --
  const int rdil[4] = {1, 2, 2, 4};
  for (int l = 0; l < 4; ++l) {
    run_conv(st, A,               2, H, W, CW[l],     Bb,               1, rdil[l], 0);
    run_conv(st, A + 2 * hwF * 64, 2, H, W, CW[4 + l], Bb + 2 * hwF * 64, 1, rdil[l], 0);
    // IN(conv) + residual + lrelu   (instance stats over all 4 samples)
    int nstat = 4 * 64;
    float* mean = (float*)aalloc(ar, nstat * 4);
    float* var  = (float*)aalloc(ar, nstat * 4);
    rn_stats<<<4, 256, 0, st>>>(Bb, mean, var, 4, H, W, 64, 64, 1);
    rn_norm_act<<<nblk((long long)4 * hwF * 64), 256, 0, st>>>(
        Bb, mean, var, A, A, 4, (long long)hwF, 64, 64, 1, 1);
  }

  // ---- share stages: conv(s1,d)+IN+act then conv(s2,d)+IN+act --------------
  auto share = [&](const bf16* x, int h, int w, int k0, int d, bf16* t, bf16* fout) {
    run_conv(st, x, 4, h, w, CW[k0], t, 1, d, 0);
    run_norm(st, ar, t, nullptr, 4, h, w, 64, 64, 1, 1);
    run_conv(st, t, 4, h, w, CW[k0 + 1], fout, 2, d, 0);
    run_norm(st, ar, fout, nullptr, 4, h / 2, w / 2, 64, 64, 1, 1);
  };
  bf16* t256 = Bb;
  bf16* f1 = (bf16*)aalloc(ar, 4 * (hwF / 4) * 64 * 2);   // (4,128,128,64)
  share(A, 256, 256, 8, 1, t256, f1);
  bf16* t128 = (bf16*)aalloc(ar, 4 * (hwF / 4) * 64 * 2);
  bf16* f2 = (bf16*)aalloc(ar, 4 * (hwF / 16) * 64 * 2);  // (4,64,64,64)
  share(f1, 128, 128, 10, 2, t128, f2);
  bf16* t64 = (bf16*)aalloc(ar, 4 * (hwF / 16) * 64 * 2);
  bf16* f3 = (bf16*)aalloc(ar, 4 * (hwF / 64) * 64 * 2);  // (4,32,32,64)
  share(f2, 64, 64, 12, 4, t64, f3);

  // ---- matcher m2 at 32x32 -------------------------------------------------
  size_t hw32 = 32 * 32, hw64 = 64 * 64, hw128 = 128 * 128;
  float* d2_32 = (float*)aalloc(ar, 2 * hw32 * 2 * 4);
  run_matcher(st, ar, CW, 21, f3, 32, 32, 2, d2_32);

  float* d2_64 = (float*)aalloc(ar, 2 * hw64 * 2 * 4);
  rn_resize2<<<nblk(2 * hw64 * 2), 256, 0, st>>>(d2_32, d2_64, 2, 32, 32, 64, 64);

  bf16* f21w = (bf16*)aalloc(ar, 2 * hw64 * 64 * 2);
  rn_warp<<<nblk(2 * hw64), 256, 0, st>>>(f2, d2_64, f21w, 2, 64, 64, 64);

  bf16* cat4m1 = (bf16*)aalloc(ar, 4 * hw64 * 64 * 2);
  (void)hipMemcpyAsync(cat4m1, f21w, 2 * hw64 * 64 * 2, hipMemcpyDeviceToDevice, st);
  (void)hipMemcpyAsync(cat4m1 + 2 * hw64 * 64, f2 + 2 * hw64 * 64,
                       2 * hw64 * 64 * 2, hipMemcpyDeviceToDevice, st);

  // ---- matcher m1 at 64x64 -------------------------------------------------
  float* d1_64 = (float*)aalloc(ar, 2 * hw64 * 2 * 4);
  run_matcher(st, ar, CW, 14, cat4m1, 64, 64, 4, d1_64);

  float* d1_128 = (float*)aalloc(ar, 2 * hw128 * 2 * 4);
  float* d2_128 = (float*)aalloc(ar, 2 * hw128 * 2 * 4);
  rn_resize2<<<nblk(2 * hw128 * 2), 256, 0, st>>>(d1_64, d1_128, 2, 64, 64, 128, 128);
  rn_resize2<<<nblk(2 * hw128 * 2), 256, 0, st>>>(d2_64, d2_128, 2, 64, 64, 128, 128);
  float* dsum = (float*)aalloc(ar, 2 * hw128 * 2 * 4);  // d1+d2 (normalized) == d4
  rn_comb<<<nblk(2 * hw128 * 2), 256, 0, st>>>(d1_128, d2_128, dsum, 2 * hw128,
                                               1.f, 1.f, 1.f, 1.f, -1e30f, 1e30f);

  bf16* f11w = (bf16*)aalloc(ar, 2 * hw128 * 64 * 2);
  rn_warp<<<nblk(2 * hw128), 256, 0, st>>>(f1, dsum, f11w, 2, 128, 128, 64);

  bf16* cat4r = (bf16*)aalloc(ar, 4 * hw128 * 64 * 2);
  (void)hipMemcpyAsync(cat4r, f11w, 2 * hw128 * 64 * 2, hipMemcpyDeviceToDevice, st);
  (void)hipMemcpyAsync(cat4r + 2 * hw128 * 64, f1 + 2 * hw128 * 64,
                       2 * hw128 * 64 * 2, hipMemcpyDeviceToDevice, st);

  // ---- refiner at 128x128 --------------------------------------------------
  float* scaleup = (float*)aalloc(ar, 2 * hw128 * 2 * 4);
  rn_comb<<<nblk(2 * hw128 * 2), 256, 0, st>>>(dsum, nullptr, scaleup, 2 * hw128,
                                               127.f, 127.f, 0.f, 0.f, -1e30f, 1e30f);
  float* dref = (float*)aalloc(ar, 2 * hw128 * 2 * 4);
  run_refiner(st, ar, CW, cat4r, 128, 128, scaleup, dref);

  float* dtmp  = (float*)aalloc(ar, 2 * hw128 * 2 * 4);
  float* dblur = (float*)aalloc(ar, 2 * hw128 * 2 * 4);
  run_blur<float>(st, dref, dtmp, dblur, 2, 128, 128, 2, 17, 5.0f, /*edge=*/1);
  float* d12 = (float*)aalloc(ar, 2 * hw128 * 2 * 4);
  rn_comb<<<nblk(2 * hw128 * 2), 256, 0, st>>>(dblur, nullptr, d12, 2 * hw128,
                                               1.f / 127.f, 1.f / 127.f, 0.f, 0.f,
                                               -1e30f, 1e30f);

  // ---- final: interp {d12, d4(=dsum), d8(=d2_128)} to 256 and emit NCHW ----
  float* outs[3] = {d12, dsum, d2_128};
  float* rtmp = (float*)aalloc(ar, 2 * hwF * 2 * 4);
  float* dout = (float*)d_out;
  for (int s = 0; s < 3; ++s) {
    rn_resize2<<<nblk(2 * hwF * 2), 256, 0, st>>>(outs[s], rtmp, 2, 128, 128, 256, 256);
    rn_out_nchw<<<nblk(2 * hwF * 2), 256, 0, st>>>(rtmp, dout + (size_t)s * 2 * 2 * hwF,
                                                   2, 256, 256);
  }
}
